// TransformerBlock_944892805660
// MI455X (gfx1250) — matrix operand order repaired and checked by reading, not compiled, not run on hardware
//
#include <hip/hip_runtime.h>
#include <hip/hip_bf16.h>
#include <math.h>

typedef __attribute__((ext_vector_type(16))) __bf16 v16bf;
typedef __attribute__((ext_vector_type(8)))  __bf16 v8bf;
typedef __attribute__((ext_vector_type(8)))  float  v8f;

#define WMMA_BF16(a, b, c) \
  __builtin_amdgcn_wmma_f32_16x16x32_bf16(false, (a), false, (b), (short)0, (c), false, false)

static __device__ inline v8f zero8() {
  v8f r;
#pragma unroll
  for (int i = 0; i < 8; ++i) r[i] = 0.0f;
  return r;
}

static __device__ inline v16bf mk16(v8bf lo, v8bf hi) {
  v16bf r;
#pragma unroll
  for (int i = 0; i < 8; ++i) { r[i] = lo[i]; r[i + 8] = hi[i]; }
  return r;
}

// A-operand (16x32, M x K): lane = row (lane&15); lanes<16 hold K 0-7 & 16-23,
// lanes>=16 hold K 8-15 & 24-31. `rowk0` points at the row's K-base.
static __device__ inline v16bf loadA(const __bf16* rowk0, int lane) {
  const int ko = (lane >> 4) * 8;
  return mk16(*(const v8bf*)(rowk0 + ko), *(const v8bf*)(rowk0 + ko + 16));
}

// B-operand (32x16, K x N): lane = column (lane&15); lanes<16 hold K 0-7 & 16-23,
// lanes>=16 hold K 8-15 & 24-31. `colk0` points at the column's K-base.
static __device__ inline v16bf loadB(const __bf16* colk0, int lane) {
  const int ko = (lane >> 4) * 8;
  return mk16(*(const v8bf*)(colk0 + ko), *(const v8bf*)(colk0 + ko + 16));
}

// ---- async global->LDS (CDNA5 GLOBAL_LOAD_ASYNC_TO_LDS_B128, ASYNCcnt) ----
static __device__ inline void async_ld16(const __bf16* ldsDst, const __bf16* gSrc) {
  unsigned l = (unsigned)(unsigned long long)
      (__attribute__((address_space(3))) const void*)ldsDst;
  asm volatile("global_load_async_to_lds_b128 %0, %1, off"
               :: "v"(l), "v"((unsigned long long)gSrc) : "memory");
}
static __device__ inline void wait_async4() {
  asm volatile("s_wait_asynccnt 0x4" ::: "memory");
}
static __device__ inline void wait_async0() {
  asm volatile("s_wait_asynccnt 0x0" ::: "memory");
}

// ---- VALU (DPP16) row reductions over 16-lane groups: xor1,xor2,xor7,xor15 ----
#if __has_builtin(__builtin_amdgcn_update_dpp)
template <int CTRL>
static __device__ inline float dpp_movf(float x) {
  int r = __builtin_amdgcn_update_dpp(0, __builtin_bit_cast(int, x),
                                      CTRL, 0xF, 0xF, true);
  return __builtin_bit_cast(float, r);
}
static __device__ inline float redmax16(float v) {
  v = fmaxf(v, dpp_movf<0xB1>(v));    // quad_perm [1,0,3,2]  (xor 1)
  v = fmaxf(v, dpp_movf<0x4E>(v));    // quad_perm [2,3,0,1]  (xor 2)
  v = fmaxf(v, dpp_movf<0x141>(v));   // row_half_mirror      (xor 7)
  v = fmaxf(v, dpp_movf<0x140>(v));   // row_mirror           (xor 15)
  return v;
}
static __device__ inline float redsum16(float v) {
  v += dpp_movf<0xB1>(v);
  v += dpp_movf<0x4E>(v);
  v += dpp_movf<0x141>(v);
  v += dpp_movf<0x140>(v);
  return v;
}
#else
static __device__ inline float redmax16(float v) {
  v = fmaxf(v, __shfl_xor(v, 1, 32));
  v = fmaxf(v, __shfl_xor(v, 2, 32));
  v = fmaxf(v, __shfl_xor(v, 4, 32));
  v = fmaxf(v, __shfl_xor(v, 8, 32));
  return v;
}
static __device__ inline float redsum16(float v) {
  v += __shfl_xor(v, 1, 32);
  v += __shfl_xor(v, 2, 32);
  v += __shfl_xor(v, 4, 32);
  v += __shfl_xor(v, 8, 32);
  return v;
}
#endif

// ---------------------------------------------------------------- utilities
__global__ void f32_to_bf16_kernel(const float* __restrict__ s, __bf16* __restrict__ d, int n) {
  int i = blockIdx.x * 256 + threadIdx.x;
  if (i < n) d[i] = (__bf16)s[i];
}

// mods[b][6144] = c[b] @ ada_w.T + ada_b   (one warp per output element)
__global__ void ada_mods_kernel(const float* __restrict__ c, const float* __restrict__ ada_w,
                                const float* __restrict__ ada_b, float* __restrict__ mods) {
  int gid = blockIdx.x * 8 + (threadIdx.x >> 5);
  if (gid >= 2 * 6144) return;
  int b = gid / 6144, row = gid % 6144;
  int lane = threadIdx.x & 31;
  const float* cv = c + (size_t)b * 1024;
  const float* wv = ada_w + (size_t)row * 1024;
  float s = 0.0f;
  for (int k = lane; k < 1024; k += 32) s += cv[k] * wv[k];
#pragma unroll
  for (int m = 16; m > 0; m >>= 1) s += __shfl_xor(s, m, 32);
  if (lane == 0) mods[gid] = s + ada_b[row];
}

// LayerNorm + adaLN modulate, f32 in -> bf16 out. One block per row (D=1024).
__global__ __launch_bounds__(256)
void ln_mod_kernel(const float* __restrict__ x, const float* __restrict__ w,
                   const float* __restrict__ mods, int scaleOff, int shiftOff,
                   __bf16* __restrict__ out) {
  const int row = blockIdx.x;           // 0..4095
  const int b = row >> 11;              // S = 2048
  const int tid = threadIdx.x;
  const float* xr = x + (size_t)row * 1024;
  __shared__ float red[256];

  float s = 0.0f;
  for (int i = tid; i < 1024; i += 256) s += xr[i];
  red[tid] = s; __syncthreads();
  for (int off = 128; off > 0; off >>= 1) { if (tid < off) red[tid] += red[tid + off]; __syncthreads(); }
  float mu = red[0] * (1.0f / 1024.0f);
  __syncthreads();

  float v = 0.0f;
  for (int i = tid; i < 1024; i += 256) { float d = xr[i] - mu; v += d * d; }
  red[tid] = v; __syncthreads();
  for (int off = 128; off > 0; off >>= 1) { if (tid < off) red[tid] += red[tid + off]; __syncthreads(); }
  float rstd = rsqrtf(red[0] * (1.0f / 1024.0f) + 1e-5f);

  const float* mrow = mods + (size_t)b * 6144;
  for (int i = tid; i < 1024; i += 256) {
    float h = (xr[i] - mu) * rstd * w[i] * (1.0f + mrow[scaleOff + i]) + mrow[shiftOff + i];
    out[(size_t)row * 1024 + i] = (__bf16)h;
  }
}

// RoPE on q,k,v (reference rotates the whole qkv tensor, v included).
// q,k -> [bh][s][64] bf16 ; v -> transposed [bh][64][s] bf16 for the PV WMMA B-operand.
__global__ void rope_kernel(const float* __restrict__ qkv, const float* __restrict__ cosb,
                            const float* __restrict__ sinb, __bf16* __restrict__ q,
                            __bf16* __restrict__ k, __bf16* __restrict__ vT) {
  int idx = blockIdx.x * 256 + threadIdx.x;          // B*S*H*64 = 2^22
  int d = idx & 63, h = (idx >> 6) & 15, s = (idx >> 10) & 2047, b = idx >> 21;
  int m = b * 2048 + s;
  const float* base = qkv + (size_t)m * 3072;
  int col = h * 64 + d;
  int dp = (d < 32) ? d + 32 : d - 32;
  float sgn = (d < 32) ? -1.0f : 1.0f;
  float cs = cosb[s * 64 + d], sn = sinb[s * 64 + d] * sgn;
  float qv = base[col] * cs        + base[h * 64 + dp] * sn;
  float kv = base[1024 + col] * cs + base[1024 + h * 64 + dp] * sn;
  float vv = base[2048 + col] * cs + base[2048 + h * 64 + dp] * sn;
  int bh = b * 16 + h;
  q[((size_t)bh * 2048 + s) * 64 + d] = (__bf16)qv;
  k[((size_t)bh * 2048 + s) * 64 + d] = (__bf16)kv;
  vT[((size_t)bh * 64 + d) * 2048 + s] = (__bf16)vv;
}

// Flash attention. Grid (S/128, B*H). 8 waves/block, 16 query rows per wave.
__global__ __launch_bounds__(256)
void attn_kernel(const __bf16* __restrict__ Q, const __bf16* __restrict__ Kt,
                 const __bf16* __restrict__ Vt, __bf16* __restrict__ O) {
  __shared__ __attribute__((aligned(16))) __bf16 Plds[8][16][32];  // wave-private tiles
  const int tid = threadIdx.x, lane = tid & 31, w = tid >> 5;
  const int bh = blockIdx.y, b = bh >> 4, h = bh & 15;
  const int q0 = blockIdx.x * 128 + w * 16;
  const int lr = lane & 15;
  const __bf16* Qp = Q + (size_t)bh * 2048 * 64;
  const __bf16* Kp = Kt + (size_t)bh * 2048 * 64;
  const __bf16* Vp = Vt + (size_t)bh * 64 * 2048;

  const int qr = q0 + lr;
  const v16bf a0 = loadA(Qp + (size_t)qr * 64, lane);        // d 0..31
  const v16bf a1 = loadA(Qp + (size_t)qr * 64 + 32, lane);   // d 32..63

  float mr[8], lsum[8];
  v8f o[4];
#pragma unroll
  for (int r = 0; r < 8; ++r) { mr[r] = -1e30f; lsum[r] = 0.0f; }
  o[0] = zero8(); o[1] = zero8(); o[2] = zero8(); o[3] = zero8();

  for (int j0 = 0; j0 < 2048; j0 += 32) {
    // scores: two 16-key n-tiles, K=64 split into two WMMAs each
    v8f c0 = zero8(), c1 = zero8();
    {
      v16bf kb0lo = loadB(Kp + (size_t)(j0 + lr) * 64, lane);
      v16bf kb0hi = loadB(Kp + (size_t)(j0 + lr) * 64 + 32, lane);
      v16bf kb1lo = loadB(Kp + (size_t)(j0 + 16 + lr) * 64, lane);
      v16bf kb1hi = loadB(Kp + (size_t)(j0 + 16 + lr) * 64 + 32, lane);
      c0 = WMMA_BF16(a0, kb0lo, c0);
      c0 = WMMA_BF16(a1, kb0hi, c0);
      c1 = WMMA_BF16(a0, kb1lo, c1);
      c1 = WMMA_BF16(a1, kb1hi, c1);
    }
    float s0[8], s1[8];
#pragma unroll
    for (int r = 0; r < 8; ++r) { s0[r] = c0[r] * 0.125f; s1[r] = c1[r] * 0.125f; }
    float corr[8], p0[8], p1[8];
#pragma unroll
    for (int r = 0; r < 8; ++r) {
      float rm = redmax16(fmaxf(s0[r], s1[r]));    // row-max (DPP, VALU-only)
      float mn = fmaxf(mr[r], rm);
      corr[r] = __expf(mr[r] - mn);
      mr[r] = mn;
      p0[r] = __expf(s0[r] - mn);
      p1[r] = __expf(s1[r] - mn);
      float sum = redsum16(p0[r] + p1[r]);         // row-sum (DPP, VALU-only)
      lsum[r] = lsum[r] * corr[r] + sum;
    }
#pragma unroll
    for (int dc = 0; dc < 4; ++dc)
#pragma unroll
      for (int r = 0; r < 8; ++r) o[dc][r] *= corr[r];

    // C-layout -> A-layout transpose through wave-private LDS (DS ops in-order per wave)
    const int rb = (lane >> 4) * 8;
#pragma unroll
    for (int r = 0; r < 8; ++r) {
      Plds[w][rb + r][lr]      = (__bf16)p0[r];
      Plds[w][rb + r][lr + 16] = (__bf16)p1[r];
    }
    v16bf pa = loadA(&Plds[w][lr][0], lane);
#pragma unroll
    for (int dc = 0; dc < 4; ++dc) {
      v16bf vb = loadB(Vp + (size_t)(dc * 16 + lr) * 2048 + j0, lane);
      o[dc] = WMMA_BF16(pa, vb, o[dc]);
    }
  }

  const int rb = (lane >> 4) * 8;
  float inv[8];
#pragma unroll
  for (int r = 0; r < 8; ++r) inv[r] = 1.0f / lsum[r];
#pragma unroll
  for (int dc = 0; dc < 4; ++dc)
#pragma unroll
    for (int r = 0; r < 8; ++r) {
      int row = b * 2048 + q0 + rb + r;
      int col = h * 64 + dc * 16 + lr;
      O[(size_t)row * 1024 + col] = (__bf16)(o[dc][r] * inv[r]);
    }
}

// GEMM: out[M][N] = A[M][K](bf16) @ W[N][K]^T(bf16), f32 accumulate.
// Double-buffered async (GLOBAL_LOAD_ASYNC_TO_LDS_B128) K-panel staging.
// epi 0: store f32. epi 1: bias + exact GELU -> bf16. epi 2: (bias?) + gate*acc + resid -> f32.
__global__ __launch_bounds__(128)
void gemm_bf16_kernel(const __bf16* __restrict__ A, const __bf16* __restrict__ W,
                      int M, int N, int K, int epi,
                      float* __restrict__ outF, __bf16* __restrict__ outB,
                      const float* __restrict__ bias, const float* __restrict__ mods,
                      int gateOff, const float* __restrict__ resid) {
  __shared__ __attribute__((aligned(16))) __bf16 As[2][64][40];
  __shared__ __attribute__((aligned(16))) __bf16 Ws[2][64][40];
  const int tid = threadIdx.x, lane = tid & 31, w = tid >> 5;
  const int n0 = blockIdx.x * 64, m0 = blockIdx.y * 64;
  const int wm = (w >> 1) * 32, wn = (w & 1) * 32;
  const int lr = lane & 15;

  // this thread's two 16B staging chunks (64 rows x 4 chunks, 128 threads)
  const int r0 = tid >> 2,          cc0 = (tid & 3) * 8;
  const int r1 = (tid + 128) >> 2,  cc1 = (tid & 3) * 8;
  const __bf16* Arow0 = A + (size_t)(m0 + r0) * K + cc0;
  const __bf16* Arow1 = A + (size_t)(m0 + r1) * K + cc1;
  const __bf16* Wrow0 = W + (size_t)(n0 + r0) * K + cc0;
  const __bf16* Wrow1 = W + (size_t)(n0 + r1) * K + cc1;

  v8f acc[2][2];
  acc[0][0] = zero8(); acc[0][1] = zero8(); acc[1][0] = zero8(); acc[1][1] = zero8();

  const int NT = K >> 5;
  // prologue: stage K-panel 0 into buffer 0
  async_ld16(&As[0][r0][cc0], Arow0);
  async_ld16(&Ws[0][r0][cc0], Wrow0);
  async_ld16(&As[0][r1][cc1], Arow1);
  async_ld16(&Ws[0][r1][cc1], Wrow1);

  for (int kt = 0; kt < NT; ++kt) {
    const int cur = kt & 1;
    if (kt + 1 < NT) {
      const int nxt = cur ^ 1, kn = (kt + 1) << 5;
      async_ld16(&As[nxt][r0][cc0], Arow0 + kn);
      async_ld16(&Ws[nxt][r0][cc0], Wrow0 + kn);
      async_ld16(&As[nxt][r1][cc1], Arow1 + kn);
      async_ld16(&Ws[nxt][r1][cc1], Wrow1 + kn);
      wait_async4();   // in-order: <=4 outstanding => panel `cur` has landed
    } else {
      wait_async0();
    }
    __syncthreads();

    v16bf af[2], bfr[2];
#pragma unroll
    for (int i = 0; i < 2; ++i) {
      af[i]  = loadA(&As[cur][wm + i * 16 + lr][0], lane);
      bfr[i] = loadB(&Ws[cur][wn + i * 16 + lr][0], lane);
    }
#pragma unroll
    for (int i = 0; i < 2; ++i)
#pragma unroll
      for (int j = 0; j < 2; ++j)
        acc[i][j] = WMMA_BF16(af[i], bfr[j], acc[i][j]);
    __syncthreads();   // protect buffer `cur` from next round's async writes
  }

  const int rb = (lane >> 4) * 8;
#pragma unroll
  for (int i = 0; i < 2; ++i)
#pragma unroll
    for (int j = 0; j < 2; ++j)
#pragma unroll
      for (int r = 0; r < 8; ++r) {
        int row = m0 + wm + i * 16 + rb + r;
        int col = n0 + wn + j * 16 + lr;
        float v = acc[i][j][r];
        size_t off = (size_t)row * N + col;
        if (epi == 0) {
          outF[off] = v;
        } else if (epi == 1) {
          v += bias[col];
          outB[off] = (__bf16)(0.5f * v * (1.0f + erff(v * 0.70710678118654752f)));
        } else {
          if (bias) v += bias[col];
          int b = row >> 11;  // S = 2048
          float g = mods[(size_t)b * 6144 + gateOff + col];
          outF[off] = g * v + resid[off];
        }
      }
}

// ---------------------------------------------------------------- launcher
static const size_t OFF_WQKV = 0;                                // 3M bf16
static const size_t OFF_WOUT = OFF_WQKV + 3145728ull * 2;        // 1M bf16
static const size_t OFF_WFC1 = OFF_WOUT + 1048576ull * 2;        // 4M bf16
static const size_t OFF_WFC2 = OFF_WFC1 + 4194304ull * 2;        // 4M bf16
static const size_t OFF_MODS = OFF_WFC2 + 4194304ull * 2;        // 12288 f32
static const size_t OFF_H    = OFF_MODS + 12288ull * 4;          // 4M bf16 (h1, reused for h2)
static const size_t OFF_QKV  = OFF_H + 4194304ull * 2;           // 12.5M f32 (reused for fc1 out bf16)
static const size_t OFF_G    = OFF_QKV;
static const size_t OFF_Q    = OFF_QKV + 4096ull * 3072 * 4;
static const size_t OFF_K    = OFF_Q + 4194304ull * 2;
static const size_t OFF_VT   = OFF_K + 4194304ull * 2;
static const size_t OFF_O    = OFF_VT + 4194304ull * 2;
static const size_t OFF_X2   = OFF_O + 4194304ull * 2;           // 4M f32

extern "C" void kernel_launch(void* const* d_in, const int* in_sizes, int n_in,
                              void* d_out, int out_size, void* d_ws, size_t ws_size,
                              hipStream_t stream) {
  (void)in_sizes; (void)n_in; (void)out_size; (void)ws_size;
  const float* x    = (const float*)d_in[0];
  const float* rcos = (const float*)d_in[1];
  const float* rsin = (const float*)d_in[2];
  const float* cv   = (const float*)d_in[3];
  const float* n1w  = (const float*)d_in[4];
  const float* qkvw = (const float*)d_in[5];
  const float* outw = (const float*)d_in[6];
  const float* n2w  = (const float*)d_in[7];
  const float* fc1w = (const float*)d_in[8];
  const float* fc1b = (const float*)d_in[9];
  const float* fc2w = (const float*)d_in[10];
  const float* fc2b = (const float*)d_in[11];
  const float* adaw = (const float*)d_in[12];
  const float* adab = (const float*)d_in[13];

  char* ws = (char*)d_ws;
  __bf16* wqkv = (__bf16*)(ws + OFF_WQKV);
  __bf16* wout = (__bf16*)(ws + OFF_WOUT);
  __bf16* wfc1 = (__bf16*)(ws + OFF_WFC1);
  __bf16* wfc2 = (__bf16*)(ws + OFF_WFC2);
  float*  mods = (float*)(ws + OFF_MODS);
  __bf16* hbf  = (__bf16*)(ws + OFF_H);
  float*  qkvf = (float*)(ws + OFF_QKV);
  __bf16* gbf  = (__bf16*)(ws + OFF_G);
  __bf16* qb   = (__bf16*)(ws + OFF_Q);
  __bf16* kb   = (__bf16*)(ws + OFF_K);
  __bf16* vtb  = (__bf16*)(ws + OFF_VT);
  __bf16* ob   = (__bf16*)(ws + OFF_O);
  float*  x2   = (float*)(ws + OFF_X2);
  float*  out  = (float*)d_out;

  // weights -> bf16 (stay resident in 192MB L2 across the GEMMs)
  f32_to_bf16_kernel<<<(3145728 + 255) / 256, 256, 0, stream>>>(qkvw, wqkv, 3145728);
  f32_to_bf16_kernel<<<(1048576 + 255) / 256, 256, 0, stream>>>(outw, wout, 1048576);
  f32_to_bf16_kernel<<<(4194304 + 255) / 256, 256, 0, stream>>>(fc1w, wfc1, 4194304);
  f32_to_bf16_kernel<<<(4194304 + 255) / 256, 256, 0, stream>>>(fc2w, wfc2, 4194304);

  ada_mods_kernel<<<1536, 256, 0, stream>>>(cv, adaw, adab, mods);

  // LN1 + modulate (scale_msa @ 1024, shift_msa @ 0)
  ln_mod_kernel<<<4096, 256, 0, stream>>>(x, n1w, mods, 1024, 0, hbf);

  // qkv = h @ qkv_w^T : 4096 x 3072 x 1024
  gemm_bf16_kernel<<<dim3(48, 64), 128, 0, stream>>>(hbf, wqkv, 4096, 3072, 1024, 0,
                                                     qkvf, nullptr, nullptr, nullptr, 0, nullptr);

  rope_kernel<<<16384, 256, 0, stream>>>(qkvf, rcos, rsin, qb, kb, vtb);

  attn_kernel<<<dim3(16, 32), 256, 0, stream>>>(qb, kb, vtb, ob);

  // x2 = gate_msa * (o @ out_w^T) + x      (gate_msa @ 2048)
  gemm_bf16_kernel<<<dim3(16, 64), 128, 0, stream>>>(ob, wout, 4096, 1024, 1024, 2,
                                                     x2, nullptr, nullptr, mods, 2048, x);

  // LN2 + modulate (scale_mlp @ 4096, shift_mlp @ 3072)
  ln_mod_kernel<<<4096, 256, 0, stream>>>(x2, n2w, mods, 4096, 3072, hbf);

  // g = gelu(h2 @ fc1_w^T + fc1_b) : 4096 x 4096 x 1024
  gemm_bf16_kernel<<<dim3(64, 64), 128, 0, stream>>>(hbf, wfc1, 4096, 4096, 1024, 1,
                                                     nullptr, gbf, fc1b, nullptr, 0, nullptr);

  // out = gate_mlp * (g @ fc2_w^T + fc2_b) + x2   (gate_mlp @ 5120)
  gemm_bf16_kernel<<<dim3(16, 64), 128, 0, stream>>>(gbf, wfc2, 4096, 1024, 4096, 2,
                                                     out, nullptr, fc2b, mods, 5120, x2);
}
